// HybridAttention_38302518346029
// MI455X (gfx1250) — matrix operand order repaired and checked by reading, not compiled, not run on hardware
//
#include <hip/hip_runtime.h>
#include <hip/hip_bf16.h>
#include <cstdint>
#include <cstddef>

// Problem constants (B=1)
constexpr int SEQ    = 2048;
constexpr int HIDDEN = 2048;
constexpr int NHEADS = 16;
constexpr int HD     = 128;

typedef __bf16 v16bf __attribute__((ext_vector_type(16)));
typedef float  v8f   __attribute__((ext_vector_type(8)));
typedef unsigned int v4u __attribute__((ext_vector_type(4)));

union Frag {
    v16bf bf;
    v4u   u[2];
};

__device__ __forceinline__ v8f wmma_bf16(const Frag& a, const Frag& b, v8f c) {
    // D = A(16x32 bf16) x B(32x16 bf16) + C(16x16 f32)
    return __builtin_amdgcn_wmma_f32_16x16x32_bf16(
        /*neg_a=*/false, a.bf, /*neg_b=*/false, b.bf,
        /*c_mod=*/(short)0, c, /*reuse_a=*/false, /*reuse_b=*/false);
}

// ---------------------------------------------------------------------------
// f32 -> bf16 elementwise convert, 8 elements per thread
// ---------------------------------------------------------------------------
__global__ void cvt_f32_to_bf16(const float* __restrict__ in,
                                __bf16* __restrict__ out, int n) {
    int i = (blockIdx.x * blockDim.x + threadIdx.x) * 8;
    if (i + 8 > n) return;
    float4 a = *(const float4*)(in + i);
    float4 b = *(const float4*)(in + i + 4);
    union { __bf16 h[8]; v4u v; } o;
    o.h[0] = (__bf16)a.x; o.h[1] = (__bf16)a.y;
    o.h[2] = (__bf16)a.z; o.h[3] = (__bf16)a.w;
    o.h[4] = (__bf16)b.x; o.h[5] = (__bf16)b.y;
    o.h[6] = (__bf16)b.z; o.h[7] = (__bf16)b.w;
    *(v4u*)(out + i) = o.v;
}

// ---------------------------------------------------------------------------
// LDS-tiled 2048x2048 bf16 transpose (for V -> V^T)
// ---------------------------------------------------------------------------
__global__ __launch_bounds__(256)
void transpose_bf16(const __bf16* __restrict__ in, __bf16* __restrict__ out) {
    __shared__ __bf16 tile[32][33];
    int x  = blockIdx.x * 32 + threadIdx.x;
    int y0 = blockIdx.y * 32;
    for (int j = threadIdx.y; j < 32; j += 8)
        tile[j][threadIdx.x] = in[(size_t)(y0 + j) * HIDDEN + x];
    __syncthreads();
    int x2 = blockIdx.y * 32 + threadIdx.x;
    int y2 = blockIdx.x * 32;
    for (int j = threadIdx.y; j < 32; j += 8)
        out[(size_t)(y2 + j) * SEQ + x2] = tile[threadIdx.x][j];
}

// ---------------------------------------------------------------------------
// C[M,N] = A[M,K] @ W[N,K]^T   (bf16 in, bf16 or f32-rounded-via-bf16 out)
// Wave tile 64x64 (4x4 WMMA subtiles): 16 WMMAs per 16 b128 loads per k-step
// (32 FLOP/byte from L2). 4 waves (2x2) -> block tile 128x128.
// A-frag (16x32 MxK): lane l (m=l&15,h=l>>4): 16B chunks at k0+8h and k0+16+8h
// B-frag (32x16 KxN): B[k][n]=W[n][k]: lane l (n=l&15): 16B chunks at k0+8h and k0+16+8h
// ---------------------------------------------------------------------------
template <bool OUT_F32>
__global__ __launch_bounds__(128)
void gemm_abt(const __bf16* __restrict__ A, const __bf16* __restrict__ W,
              void* __restrict__ Cv, int M, int N, int K) {
    const int lane = threadIdx.x & 31;
    const int wave = threadIdx.x >> 5;
    const int t  = lane & 15;
    const int hh = lane >> 4;
    const int m0 = blockIdx.y * 128 + (wave >> 1) * 64;
    const int n0 = blockIdx.x * 128 + (wave & 1) * 64;

    const __bf16* arow[4];
    const __bf16* brow[4];
#pragma unroll
    for (int i = 0; i < 4; ++i) {
        arow[i] = A + (size_t)(m0 + 16 * i + t) * K;
        brow[i] = W + (size_t)(n0 + 16 * i + t) * K;
    }

    v8f acc[4][4] = {};
    for (int k0 = 0; k0 < K; k0 += 32) {
        // global_prefetch_b8 one line ahead (speculative; OOB silently dropped)
        __builtin_prefetch(arow[0] + k0 + 64, 0, 3);
        __builtin_prefetch(brow[0] + k0 + 64, 0, 3);
        Frag a[4], b[4];
#pragma unroll
        for (int i = 0; i < 4; ++i) {
            const __bf16* ap = arow[i] + k0 + 8 * hh;
            a[i].u[0] = *(const v4u*)ap;
            a[i].u[1] = *(const v4u*)(ap + 16);
            const __bf16* bp = brow[i] + k0 + 8 * hh;
            b[i].u[0] = *(const v4u*)bp;
            b[i].u[1] = *(const v4u*)(bp + 16);
        }
#pragma unroll
        for (int i = 0; i < 4; ++i)
#pragma unroll
            for (int j = 0; j < 4; ++j)
                acc[i][j] = wmma_bf16(a[i], b[j], acc[i][j]);
    }
    // C/D layout: lane l holds rows (r + 8*(l>>4)), col (l&15), r=0..7
#pragma unroll
    for (int i = 0; i < 4; ++i)
#pragma unroll
        for (int j = 0; j < 4; ++j)
#pragma unroll
            for (int r = 0; r < 8; ++r) {
                size_t m = (size_t)(m0 + 16 * i + r + 8 * hh);
                size_t n = (size_t)(n0 + 16 * j + t);
                float v = acc[i][j][r];
                if (OUT_F32) ((float*)Cv)[m * N + n] = (float)(__bf16)v;
                else         ((__bf16*)Cv)[m * N + n] = (__bf16)v;
            }
}

// ---------------------------------------------------------------------------
// Flash attention: grid(SEQ/64, NHEADS, 2[self|cross]), 128 threads (4 waves).
// Each wave owns a 16-query tile; streams keys in chunks of 32.
// ---------------------------------------------------------------------------
__global__ __launch_bounds__(128)
void attn_kernel(const __bf16* __restrict__ Qs, const __bf16* __restrict__ Qc,
                 const __bf16* __restrict__ Km, const __bf16* __restrict__ Vt,
                 const float* __restrict__ mask, __bf16* __restrict__ Oflat) {
    const int lane = threadIdx.x & 31;
    const int wave = threadIdx.x >> 5;
    const int t  = lane & 15;
    const int hh = lane >> 4;
    const int head  = blockIdx.y;
    const int blk   = blockIdx.z;                 // 0=self, 1=cross
    const int qbase = blockIdx.x * 64 + wave * 16;
    const float scale = 0.08838834764831845f;     // 128^-0.5

    __shared__ __align__(16) __bf16 sm[4][16][32]; // per-wave P staging tile

    const __bf16* Qmat = blk ? Qc : Qs;

    // Q fragments for the whole 16x128 tile (4 k-blocks of 32)
    Frag qf[4];
#pragma unroll
    for (int kb = 0; kb < 4; ++kb) {
        const __bf16* ap = Qmat + (size_t)(qbase + t) * HIDDEN + head * HD + 32 * kb + 8 * hh;
        qf[kb].u[0] = *(const v4u*)ap;
        qf[kb].u[1] = *(const v4u*)(ap + 16);
    }

    v8f o[8] = {};            // 8 feature subtiles of 16 -> 16x128 output
    float mrow[8], lrow[8];   // running max / sum for rows (r + 8*hh)
#pragma unroll
    for (int r = 0; r < 8; ++r) { mrow[r] = -3.0e38f; lrow[r] = 0.f; }

    for (int kc = 0; kc < SEQ; kc += 32) {
        // ---- scores: S = Q @ K^T for 32 keys (two 16-key accumulators) ----
        v8f s0 = {}, s1 = {};
#pragma unroll
        for (int kb = 0; kb < 4; ++kb) {
            Frag b0, b1;
            const __bf16* bp0 = Km + (size_t)(kc + t) * HIDDEN + head * HD + 32 * kb + 8 * hh;
            const __bf16* bp1 = bp0 + (size_t)16 * HIDDEN;
            b0.u[0] = *(const v4u*)bp0; b0.u[1] = *(const v4u*)(bp0 + 16);
            b1.u[0] = *(const v4u*)bp1; b1.u[1] = *(const v4u*)(bp1 + 16);
            s0 = wmma_bf16(qf[kb], b0, s0);
            s1 = wmma_bf16(qf[kb], b1, s1);
        }
        // ---- online softmax (rows live as: VGPR r <-> row r+8*hh, lanes = keys) ----
#pragma unroll
        for (int r = 0; r < 8; ++r) {
            int q = qbase + r + 8 * hh;
            float v0 = s0[r] * scale + mask[(size_t)q * SEQ + kc + t];
            float v1 = s1[r] * scale + mask[(size_t)q * SEQ + kc + 16 + t];
            float cm = fmaxf(v0, v1);
#pragma unroll
            for (int d = 1; d < 16; d <<= 1) cm = fmaxf(cm, __shfl_xor(cm, d, 32));
            float nm = fmaxf(mrow[r], cm);
            float p0 = __expf(v0 - nm);
            float p1 = __expf(v1 - nm);
            float al = __expf(mrow[r] - nm);
            float rs = p0 + p1;
#pragma unroll
            for (int d = 1; d < 16; d <<= 1) rs += __shfl_xor(rs, d, 32);
            lrow[r] = lrow[r] * al + rs;
            mrow[r] = nm;
#pragma unroll
            for (int nf = 0; nf < 8; ++nf) o[nf][r] *= al;
            // stage P (bf16) row-major [16][32] for A-frag reload
            sm[wave][r + 8 * hh][t]      = (__bf16)p0;
            sm[wave][r + 8 * hh][t + 16] = (__bf16)p1;
        }
        __syncthreads();
        // ---- reload P in A-fragment layout (16x32 MxK) ----
        Frag pf;
        pf.u[0] = *(const v4u*)&sm[wave][t][8 * hh];
        pf.u[1] = *(const v4u*)&sm[wave][t][16 + 8 * hh];
        // ---- O += P @ V   (V^T gives contiguous B-frags) ----
#pragma unroll
        for (int nf = 0; nf < 8; ++nf) {
            Frag vf;
            const __bf16* vp = Vt + (size_t)(head * HD + nf * 16 + t) * SEQ + kc + 8 * hh;
            vf.u[0] = *(const v4u*)vp;
            vf.u[1] = *(const v4u*)(vp + 16);
            o[nf] = wmma_bf16(pf, vf, o[nf]);
        }
        __syncthreads();
    }
    // ---- normalize and write into interleaved out_flat: row 2q+blk, col head*128+d ----
#pragma unroll
    for (int nf = 0; nf < 8; ++nf)
#pragma unroll
        for (int r = 0; r < 8; ++r) {
            int q = qbase + r + 8 * hh;
            size_t row = (size_t)2 * q + blk;
            size_t col = (size_t)(head * HD + nf * 16 + t);
            Oflat[row * HIDDEN + col] = (__bf16)(o[nf][r] / lrow[r]);
        }
}

// ---------------------------------------------------------------------------
extern "C" void kernel_launch(void* const* d_in, const int* in_sizes, int n_in,
                              void* d_out, int out_size, void* d_ws, size_t ws_size,
                              hipStream_t stream) {
    (void)in_sizes; (void)n_in; (void)out_size; (void)ws_size;
    const float* x_self  = (const float*)d_in[0];
    const float* x_cross = (const float*)d_in[1];
    const float* mask    = (const float*)d_in[2];
    const float* Wq      = (const float*)d_in[3];
    const float* Wk      = (const float*)d_in[4];
    const float* Wv      = (const float*)d_in[5];
    const float* Wo      = (const float*)d_in[6];
    float* out = (float*)d_out;

    const size_t NE = (size_t)SEQ * HIDDEN;  // 4M elements, 8MB as bf16
    char* ws = (char*)d_ws;
    size_t off = 0;
    auto grab = [&](size_t elems) {
        __bf16* p = (__bf16*)(ws + off);
        off += elems * sizeof(__bf16);
        return p;
    };
    __bf16* Xs  = grab(NE);
    __bf16* Xc  = grab(NE);
    __bf16* Wqb = grab(NE);
    __bf16* Wkb = grab(NE);
    __bf16* Wvb = grab(NE);
    __bf16* Wob = grab(NE);
    __bf16* Q_s = grab(NE);
    __bf16* Q_c = grab(NE);
    __bf16* Kb  = grab(NE);
    __bf16* Vb  = grab(NE);
    __bf16* Vt  = grab(NE);
    __bf16* Ofl = grab(2 * NE);   // total ws use: 104 MB

    // 1) bf16 conversions
    const int n = (int)NE;
    const int cgrid = n / (256 * 8);
    cvt_f32_to_bf16<<<cgrid, 256, 0, stream>>>(x_self,  Xs,  n);
    cvt_f32_to_bf16<<<cgrid, 256, 0, stream>>>(x_cross, Xc,  n);
    cvt_f32_to_bf16<<<cgrid, 256, 0, stream>>>(Wq,      Wqb, n);
    cvt_f32_to_bf16<<<cgrid, 256, 0, stream>>>(Wk,      Wkb, n);
    cvt_f32_to_bf16<<<cgrid, 256, 0, stream>>>(Wv,      Wvb, n);
    cvt_f32_to_bf16<<<cgrid, 256, 0, stream>>>(Wo,      Wob, n);

    // 2) projections (bf16 WMMA GEMMs, C = X @ W^T), block tile 128x128
    dim3 gp(HIDDEN / 128, SEQ / 128);
    gemm_abt<false><<<gp, 128, 0, stream>>>(Xs, Wqb, Q_s, SEQ, HIDDEN, HIDDEN);
    gemm_abt<false><<<gp, 128, 0, stream>>>(Xc, Wqb, Q_c, SEQ, HIDDEN, HIDDEN);
    gemm_abt<false><<<gp, 128, 0, stream>>>(Xs, Wkb, Kb,  SEQ, HIDDEN, HIDDEN);
    gemm_abt<false><<<gp, 128, 0, stream>>>(Xs, Wvb, Vb,  SEQ, HIDDEN, HIDDEN);

    // 3) V^T for contiguous AV B-fragments
    transpose_bf16<<<dim3(HIDDEN / 32, SEQ / 32), dim3(32, 8), 0, stream>>>(Vb, Vt);

    // 4) attention (self + cross share K/V), writes interleaved out_flat
    attn_kernel<<<dim3(SEQ / 64, NHEADS, 2), 128, 0, stream>>>(Q_s, Q_c, Kb, Vt, mask, Ofl);

    // 5) output projection: out[4096,2048] f32 = out_flat @ Wo^T
    gemm_abt<true><<<dim3(HIDDEN / 128, (2 * SEQ) / 128), 128, 0, stream>>>(
        Ofl, Wob, out, 2 * SEQ, HIDDEN, HIDDEN);
}
